// SpikingCNN_84524956386050
// MI455X (gfx1250) — matrix operand order repaired and checked by reading, not compiled, not run on hardware
//
#include <hip/hip_runtime.h>

typedef __attribute__((ext_vector_type(16))) _Float16 v16h;
typedef __attribute__((ext_vector_type(8)))  float    v8f;

#define BETA   0.9f
#define THR    1.0f
#define NSTEPS 10
#define IMG    256      // 16x16 pixels per sample
#define KTOT   2048     // 8 channels * 256
#define CHUNKS 64       // 2048 / 32 (WMMA K)

// One wave = 16 samples (WMMA M).  4 waves / block = 64 samples.
__global__ __launch_bounds__(128) void snn_fused(
    const float* __restrict__ x, const float* __restrict__ conv_w,
    const float* __restrict__ conv_b, const float* __restrict__ fc_w,
    const float* __restrict__ fc_b, float* __restrict__ out)
{
  extern __shared__ float xs[];              // [64][256] staged input images
  const int tid  = threadIdx.x;
  const int lane = tid & 31;
  const int wave = tid >> 5;
  const int hi   = (lane >> 4) & 1;          // lanes 16-31 -> high K half
  const int m    = lane & 15;                // A row (sample in tile) / B,C col
  const int blockBase = blockIdx.x * 64;

  // ---- cooperative b128 load of 64 sample images into LDS ----
  {
    const float4* gx = reinterpret_cast<const float4*>(x + (size_t)blockBase * IMG);
    float4* sx = reinterpret_cast<float4*>(xs);
    #pragma unroll
    for (int i = 0; i < 32; ++i) sx[tid + i * 128] = gx[tid + i * 128];
  }
  __syncthreads();

  const float* img = xs + (wave * 16 + m) * IMG;   // this lane's sample image
  const float  fcb = (m < 2) ? fc_b[m] : 0.0f;

  v8f zero = {0.f,0.f,0.f,0.f,0.f,0.f,0.f,0.f};
  v8f C[NSTEPS];
  #pragma unroll
  for (int t = 0; t < NSTEPS; ++t) C[t] = zero;

  for (int chunk = 0; chunk < CHUNKS; ++chunk) {
    const int   c   = chunk >> 3;            // conv channel (uniform in wave)
    const int   h0  = (chunk & 7) << 1;      // first of the 2 image rows
    const float bc  = conv_b[c];
    const float* cw = conv_w + c * 9;

    // ---- B operand: fc_w 32x16 tile, only cols 0/1 nonzero ----
    v16h bmat;
    {
      const int kb = chunk * 32 + (hi ? 8 : 0);
      #pragma unroll
      for (int j = 0; j < 16; ++j) {
        float wv = (m < 2) ? fc_w[m * KTOT + kb + j + ((j >= 8) ? 8 : 0)] : 0.0f;
        bmat[j] = (_Float16)wv;
      }
    }

    // ---- conv 3x3 (SAME, zero pad): this lane's 16 cur1 elements,
    //      indexed per the 16-bit A-matrix VGPR layout ----
    float cur1[16], mem1[16];
    #pragma unroll
    for (int j = 0; j < 16; ++j) {
      const int K  = j + ((j >= 8) ? 8 : 0) + (hi ? 8 : 0);
      const int hh = h0 + (K >> 4);
      const int ww = K & 15;
      float acc = bc;
      #pragma unroll
      for (int dy = 0; dy < 3; ++dy) {
        const int hy = hh + dy - 1;
        #pragma unroll
        for (int dx = 0; dx < 3; ++dx) {
          const int wx = ww + dx - 1;
          const bool ok = ((unsigned)hy < 16u) && ((unsigned)wx < 16u);
          const float xv = ok ? img[hy * 16 + wx] : 0.0f;
          acc = fmaf(xv, cw[dy * 3 + dx], acc);
        }
      }
      cur1[j] = acc;
      mem1[j] = 0.0f;
    }

    // ---- 10-step leaky integrate-and-fire, spikes -> f16 A, WMMA per step ----
    #pragma unroll
    for (int t = 0; t < NSTEPS; ++t) {
      v16h amat;
      #pragma unroll
      for (int j = 0; j < 16; ++j) {
        const float r = (mem1[j] > THR) ? THR : 0.0f;     // reset from prev mem
        mem1[j] = fmaf(BETA, mem1[j], cur1[j]) - r;
        amat[j] = (mem1[j] > THR) ? (_Float16)1.0f : (_Float16)0.0f;
      }
      C[t] = __builtin_amdgcn_wmma_f32_16x16x32_f16(
          false, amat, false, bmat, (short)0, C[t], false, false);
    }
  }

  // ---- output-layer recurrence directly on the C-matrix layout ----
  float mem2[8], accv[8];
  #pragma unroll
  for (int v = 0; v < 8; ++v) { mem2[v] = 0.0f; accv[v] = 0.0f; }
  #pragma unroll
  for (int t = 0; t < NSTEPS; ++t) {
    #pragma unroll
    for (int v = 0; v < 8; ++v) {
      const float cur2 = C[t][v] + fcb;
      const float r = (mem2[v] > THR) ? THR : 0.0f;
      mem2[v] = fmaf(BETA, mem2[v], cur2) - r;
      accv[v] += (mem2[v] > THR) ? 1.0f : 0.0f;
    }
  }

  if (m < 2) {   // C layout: lane (m<2) col m, VGPR v -> sample v + 8*hi
    #pragma unroll
    for (int v = 0; v < 8; ++v) {
      const int b = blockBase + wave * 16 + v + (hi ? 8 : 0);
      out[(size_t)b * 2 + m] = accv[v];
    }
  }
}

extern "C" void kernel_launch(void* const* d_in, const int* in_sizes, int n_in,
                              void* d_out, int out_size, void* d_ws, size_t ws_size,
                              hipStream_t stream) {
  const float* x      = (const float*)d_in[0];
  const float* conv_w = (const float*)d_in[1];
  const float* conv_b = (const float*)d_in[2];
  const float* fc_w   = (const float*)d_in[3];
  const float* fc_b   = (const float*)d_in[4];
  float* out = (float*)d_out;
  const int B = in_sizes[0] / IMG;           // 65536
  const int blocks = B / 64;
  const size_t lds = 64 * IMG * sizeof(float);   // 64 KB dynamic LDS
  hipLaunchKernelGGL(snn_fused, dim3(blocks), dim3(128), lds, stream,
                     x, conv_w, conv_b, fc_w, fc_b, out);
}
